// CombinedLoss_1288490188968
// MI455X (gfx1250) — matrix operand order repaired and checked by reading, not compiled, not run on hardware
//
#include <hip/hip_runtime.h>
#include <hip/hip_bf16.h>
#include <stdint.h>

// CombinedLoss for MI455X (gfx1250, wave32).
// B=8, C=20, N=131072.  alpha=1 (CE), beta=1 (Lovasz), gamma=0.5 (Dice), eps=1e-6.

typedef __attribute__((ext_vector_type(16))) _Float16 v16h;
typedef __attribute__((ext_vector_type(8)))  float    v8f;

#define B_   8
#define C_   20
#define N_   131072
#define SEGS (B_ * C_)
#define EPS_ 1e-6f

__device__ __forceinline__ float waveSum(float v) {
#pragma unroll
  for (int o = 16; o > 0; o >>= 1) v += __shfl_down(v, o, 32);
  return v;
}

// gfx1250 async copy: 16B/lane global -> LDS, tracked by ASYNCcnt.
__device__ __forceinline__ void asyncLoadB128(unsigned ldsOff, const void* gptr) {
  unsigned long long ga = (unsigned long long)(uintptr_t)gptr;
  asm volatile("global_load_async_to_lds_b128 %0, %1, off"
               :: "v"(ldsOff), "v"(ga) : "memory");
}
__device__ __forceinline__ void waitAsync0() {
  asm volatile("s_wait_asynccnt 0x0" ::: "memory");
}

// ---------------------------------------------------------------------------
// Kernel 0: zero the scalar accumulators (ws is poisoned; must re-init per call)
// ---------------------------------------------------------------------------
__global__ void zero_acc_kernel(float* __restrict__ acc) {
  int i = blockIdx.x * blockDim.x + threadIdx.x;
  if (i < 1 + 3 * SEGS) acc[i] = 0.0f;
}

// ---------------------------------------------------------------------------
// Kernel 1: fused softmax pass.
//   Per (b,n) column of 20 logits: log-softmax, CE nll, per-(b,c) Dice sums
//   (sum p, fg count, sum p*fg), and the packed sort key for Lovasz:
//       key = ~((float_bits(err) << 1) | fg)        (err in [0,1) -> fits)
//   Ascending sort of key == descending sort of err; err & fg recoverable.
// ---------------------------------------------------------------------------
__global__ void fused_softmax_kernel(const float* __restrict__ logits,
                                     const int*   __restrict__ target,
                                     unsigned*    __restrict__ keys,
                                     float*       __restrict__ acc) {
  __shared__ float    ldsP[C_];
  __shared__ float    ldsPF[C_];
  __shared__ unsigned ldsFG[C_];
  __shared__ float    ldsCE;

  const int tid = threadIdx.x;
  if (tid < C_) { ldsP[tid] = 0.0f; ldsPF[tid] = 0.0f; ldsFG[tid] = 0u; }
  if (tid == 0) ldsCE = 0.0f;
  __syncthreads();

  const int n = blockIdx.x * blockDim.x + tid;
  const int b = blockIdx.y;
  const int t = target[(size_t)b * N_ + n];
  const float* lp = logits + (size_t)b * C_ * N_ + n;

  float x[C_];
  float m = -3.4e38f;
#pragma unroll
  for (int c = 0; c < C_; c++) { x[c] = lp[(size_t)c * N_]; m = fmaxf(m, x[c]); }

  float s = 0.0f, xt = 0.0f;
#pragma unroll
  for (int c = 0; c < C_; c++) {
    if (c == t) xt = x[c];
    x[c] = __expf(x[c] - m);
    s += x[c];
  }
  const float lse = m + __logf(s);
  const float inv = 1.0f / s;
  const float nll = lse - xt;          // -log p_target
  const unsigned lane = tid & 31;
  float pt = 0.0f;

#pragma unroll
  for (int c = 0; c < C_; c++) {
    float p = x[c] * inv;
    unsigned fg = (c == t) ? 1u : 0u;
    if (fg) pt = p;
    float err = fg ? (1.0f - p) : p;                 // == |fg - p|
    unsigned key = ~((__float_as_uint(err) << 1) | fg);
    keys[((size_t)(b * C_ + c)) * N_ + n] = key;

    float wp = waveSum(p);
    unsigned long long bal = __ballot(c == t);       // wave32 ballot
    if (lane == 0) {
      atomicAdd(&ldsP[c], wp);                       // ds_add_f32
      unsigned cnt = (unsigned)__popcll(bal);
      if (cnt) atomicAdd(&ldsFG[c], cnt);            // ds_add_u32
    }
  }
  atomicAdd(&ldsPF[t], pt);                          // sum p*fg hits only c==t
  float wn = waveSum(nll);
  if (lane == 0) atomicAdd(&ldsCE, wn);
  __syncthreads();

  if (tid < C_) {
    atomicAdd(&acc[1 + b * C_ + tid],             ldsP[tid]);
    atomicAdd(&acc[1 + SEGS + b * C_ + tid],      (float)ldsFG[tid]);
    atomicAdd(&acc[1 + 2 * SEGS + b * C_ + tid],  ldsPF[tid]);
  }
  if (tid == 0) atomicAdd(&acc[0], ldsCE);
}

// ---------------------------------------------------------------------------
// Kernel 2: segmented LSD radix sort (ascending on inverted keys).
//   One workgroup per (b,c) segment of 131072 keys; 4 passes of 8 bits,
//   global ping-pong (1 MB working set per WG -> L2-resident on 192MB L2).
//   Equal keys are fully interchangeable, so unstable LDS-atomic scatter is OK.
//   B128 source reads; wave-parallel 256-bin exclusive scan.
// ---------------------------------------------------------------------------
__global__ void seg_radix_sort_kernel(unsigned* __restrict__ keysA,
                                      unsigned* __restrict__ keysB) {
  __shared__ unsigned hist[256];
  __shared__ unsigned offs[256];
  const int seg = blockIdx.x;
  unsigned* src = keysA + (size_t)seg * N_;
  unsigned* dst = keysB + (size_t)seg * N_;
  const int tid  = threadIdx.x;   // 256 threads
  const int lane = tid & 31;

  for (int pass = 0; pass < 4; pass++) {
    const int sh = pass * 8;
    hist[tid] = 0u;
    __syncthreads();

    const uint4* s4 = (const uint4*)src;
    for (int i = tid; i < N_ / 4; i += 256) {
      __builtin_prefetch(s4 + i + 1024, 0, 0);       // global_prefetch_b8
      uint4 v = s4[i];
      atomicAdd(&hist[(v.x >> sh) & 255u], 1u);      // ds_add_u32
      atomicAdd(&hist[(v.y >> sh) & 255u], 1u);
      atomicAdd(&hist[(v.z >> sh) & 255u], 1u);
      atomicAdd(&hist[(v.w >> sh) & 255u], 1u);
    }
    __syncthreads();

    // wave-parallel exclusive scan of 256 bins: 8 bins/lane on wave 0
    if (tid < 32) {
      unsigned h[8], lexc[8], lsum = 0;
#pragma unroll
      for (int j = 0; j < 8; j++) h[j] = hist[tid * 8 + j];
#pragma unroll
      for (int j = 0; j < 8; j++) { lexc[j] = lsum; lsum += h[j]; }
      unsigned incl = lsum;
#pragma unroll
      for (int o = 1; o < 32; o <<= 1) {
        unsigned tv = __shfl_up(incl, o, 32);
        if (lane >= o) incl += tv;
      }
      unsigned exclw = incl - lsum;
#pragma unroll
      for (int j = 0; j < 8; j++) offs[tid * 8 + j] = exclw + lexc[j];
    }
    __syncthreads();

    for (int i = tid; i < N_ / 4; i += 256) {
      uint4 v = s4[i];
      unsigned k, d, pos;
      k = v.x; d = (k >> sh) & 255u; pos = atomicAdd(&offs[d], 1u); dst[pos] = k;
      k = v.y; d = (k >> sh) & 255u; pos = atomicAdd(&offs[d], 1u); dst[pos] = k;
      k = v.z; d = (k >> sh) & 255u; pos = atomicAdd(&offs[d], 1u); dst[pos] = k;
      k = v.w; d = (k >> sh) & 255u; pos = atomicAdd(&offs[d], 1u); dst[pos] = k;
    }
    __syncthreads();
    unsigned* tmp = src; src = dst; dst = tmp;
  }
  // 4 passes: A->B->A->B->A  => sorted result lands back in keysA
}

// ---------------------------------------------------------------------------
// Kernel 3: Lovasz reduction over sorted segments.
//   Async double-buffered pipeline: global_load_async_to_lds_b128 stages the
//   next 8KB tile while the current tile computes (ASYNCcnt + barrier handoff).
//   Inclusive prefix sum of fg via WMMA tensor-core scan:
//     chunk of 256 elems laid out as 16x16 (i = r + 16c); D = L x F with
//     v_wmma_f32_16x16x32_f16 gives within-column inclusive sums (f32 exact).
//   Cross-column: width-16 shuffle scan of column totals (D row 15 = vgpr7,
//   lanes 16..31). Cross-wave: LDS. Cross-tile: register carry.
//   Each element computes its grad term locally:
//     J(i,cf) = 1 - (G-cf)/(G + i+1 - cf);  grad_i = J(i,cf_i) - J(i-1, cf_i - fg_i)
// ---------------------------------------------------------------------------
__global__ void lovasz_scan_kernel(const unsigned* __restrict__ keys,
                                   const float*    __restrict__ acc,
                                   float*          __restrict__ loss_bc) {
  __shared__ __align__(16) unsigned lbuf[2][2048];   // double-buffered key tiles
  __shared__ float ldsTot[8];
  __shared__ float ldsRed[8];
  const int seg = blockIdx.x;
  const unsigned* sk = keys + (size_t)seg * N_;
  const float G = acc[1 + SEGS + seg];       // total fg count (gts)

  const int tid  = threadIdx.x;              // 256 threads = 8 waves
  const int wave = tid >> 5;
  const int lid  = tid & 31;
  const int col  = lid & 15;
  const int half = lid >> 4;
  const int T    = N_ / 2048;                // 64 tiles

  // A = 16x16 lower-triangular ones (K<=M), zero-padded to K=32.
  // A k order used here (contiguous, same for B): lanes 0-15 (M=lid): elems 0..15 = K 0..15; lanes 16-31
  // (M=lid-16): elems 0..15 = K 16..31 -> zero.
  v16h amat;
#pragma unroll
  for (int j = 0; j < 16; j++) amat[j] = (_Float16)0.0f;
  if (half == 0) {
#pragma unroll
    for (int j = 0; j < 16; j++) amat[j] = (_Float16)((j <= lid) ? 1.0f : 0.0f);
  }

  v8f czero;
#pragma unroll
  for (int j = 0; j < 8; j++) czero[j] = 0.0f;

  // Prologue: async-stage tile 0 into slot 0 (2 x B128 per thread).
  {
    unsigned i0 = (unsigned)tid, i1 = (unsigned)tid + 256u;
    asyncLoadB128((unsigned)(uintptr_t)&lbuf[0][4 * i0], sk + 4 * i0);
    asyncLoadB128((unsigned)(uintptr_t)&lbuf[0][4 * i1], sk + 4 * i1);
  }

  float carry = 0.0f;
  float loss  = 0.0f;

  for (int tile = 0; tile < T; ++tile) {
    const int slot = tile & 1;
    waitAsync0();                  // this wave's tile data landed in LDS
    __syncthreads();               // every wave's tile data visible

    if (tile + 1 < T) {            // prefetch next tile into the other slot
      const unsigned* nsrc = sk + (tile + 1) * 2048;
      unsigned i0 = (unsigned)tid, i1 = (unsigned)tid + 256u;
      asyncLoadB128((unsigned)(uintptr_t)&lbuf[slot ^ 1][4 * i0], nsrc + 4 * i0);
      asyncLoadB128((unsigned)(uintptr_t)&lbuf[slot ^ 1][4 * i1], nsrc + 4 * i1);
    }

    const int chunkBase = tile * 2048 + wave * 256;
    unsigned k0[16];
    v16h bmat;
#pragma unroll
    for (int j = 0; j < 16; j++) bmat[j] = (_Float16)0.0f;

    if (half == 0) {
      // lane col holds the full column: elements r=0..15 at i = 16*col + r
      const uint4* kp = (const uint4*)&lbuf[slot][wave * 256 + 16 * col];
#pragma unroll
      for (int q = 0; q < 4; q++) {
        uint4 v = kp[q];
        k0[4*q+0] = v.x; k0[4*q+1] = v.y; k0[4*q+2] = v.z; k0[4*q+3] = v.w;
      }
#pragma unroll
      for (int r = 0; r < 16; r++) {
        unsigned fg = (~k0[r]) & 1u;
        bmat[r] = (_Float16)(float)fg;   // B: lanes 0-15 carry K=0..15; lanes 16-31 zero
      }
    } else {
      // upper-half lanes fetch r=8..15 of their column from LDS (for D layout)
      const uint4* kp = (const uint4*)&lbuf[slot][wave * 256 + 16 * col + 8];
#pragma unroll
      for (int q = 0; q < 2; q++) {
        uint4 v = kp[q];
        k0[4*q+0] = v.x; k0[4*q+1] = v.y; k0[4*q+2] = v.z; k0[4*q+3] = v.w;
      }
    }

    // D[r][c] = sum_{r'<=r} F[r'][c]  (within-column inclusive scan, f32 exact)
    v8f d = __builtin_amdgcn_wmma_f32_16x16x32_f16(false, amat, false, bmat,
                                                   (short)0, czero, false, false);

    // column totals: D[15][c] lives in vgpr7 on lanes 16..31
    float cs = __shfl(d[7], 16 + col, 32);
    float incl = cs;
#pragma unroll
    for (int off = 1; off < 16; off <<= 1) {
      float tv = __shfl_up(incl, off, 16);
      if (col >= off) incl += tv;
    }
    float excl = incl - cs;
    float chunkTotal = __shfl(incl, 15, 16);

    __syncthreads();
    if (lid == 0) ldsTot[wave] = chunkTotal;
    __syncthreads();
    float waveOff = 0.0f, tileTot = 0.0f;
#pragma unroll
    for (int w = 0; w < 8; w++) {
      float v2 = ldsTot[w];
      tileTot += v2;
      if (w < wave) waveOff += v2;
    }
    const float base = carry + waveOff + excl;
    carry += tileTot;

    // Per-element Lovasz terms. d[j] maps to r = j + 8*half of column `col`;
    // k0[j] holds that element's key in both halves.
#pragma unroll
    for (int j = 0; j < 8; j++) {
      int r = j + 8 * half;
      unsigned invk = ~k0[j];
      float fg  = (float)(invk & 1u);
      float err = __uint_as_float(invk >> 1);
      float cf  = base + d[j];               // inclusive cumsum(fg) at global position
      int pos   = chunkBase + 16 * col + r;
      float fpos = (float)pos;
      float Ji  = 1.0f - (G - cf) / (G + (fpos + 1.0f) - cf);
      float cfp = cf - fg;
      float Jp  = (pos == 0) ? 0.0f
                             : (1.0f - (G - cfp) / (G + fpos - cfp));
      loss += err * (Ji - Jp);
    }
  }

  float ws = waveSum(loss);
  __syncthreads();
  if (lid == 0) ldsRed[wave] = ws;
  __syncthreads();
  if (tid == 0) {
    float tot = 0.0f;
#pragma unroll
    for (int w = 0; w < 8; w++) tot += ldsRed[w];
    loss_bc[seg] = tot;
  }
}

// ---------------------------------------------------------------------------
// Kernel 4: final scalar combine (160 values -> 1 float)
// ---------------------------------------------------------------------------
__global__ void combine_kernel(const float* __restrict__ acc,
                               const float* __restrict__ loss_bc,
                               float*       __restrict__ out) {
  if (threadIdx.x == 0 && blockIdx.x == 0) {
    const float* sp  = acc + 1;
    const float* sfg = acc + 1 + SEGS;
    const float* spf = acc + 1 + 2 * SEGS;
    float ce = acc[0] / (float)((size_t)B_ * N_);
    float ds = 0.0f;
    for (int s = 0; s < SEGS; s++)
      ds += (2.0f * spf[s] + EPS_) / (sp[s] + sfg[s] + EPS_);
    float dice_loss = 1.0f - ds / (float)SEGS;
    float lov = 0.0f;
    for (int b = 0; b < B_; b++) {
      float np = 0.0f, ls = 0.0f;
      for (int c = 0; c < C_; c++) {
        int s = b * C_ + c;
        if (sfg[s] > 0.0f) { np += 1.0f; ls += loss_bc[s]; }
      }
      lov += (np > 0.0f) ? (ls / np) : 0.0f;
    }
    lov /= (float)B_;
    out[0] = 1.0f * ce + 1.0f * lov + 0.5f * dice_loss;
  }
}

// ---------------------------------------------------------------------------
extern "C" void kernel_launch(void* const* d_in, const int* in_sizes, int n_in,
                              void* d_out, int out_size, void* d_ws, size_t ws_size,
                              hipStream_t stream) {
  const float* logits = (const float*)d_in[0];
  const int*   target = (const int*)d_in[1];

  const size_t keyElems = (size_t)B_ * C_ * N_;      // 20.97M keys (80 MB)
  unsigned* keysA  = (unsigned*)d_ws;
  unsigned* keysB  = keysA + keyElems;
  float*    acc    = (float*)(keysB + keyElems);     // [0]=ce, [1..]=sum_p, fg, pfg
  float*    lossBC = acc + 1 + 3 * SEGS;
  float*    out    = (float*)d_out;

  hipLaunchKernelGGL(zero_acc_kernel, dim3(3), dim3(256), 0, stream, acc);
  hipLaunchKernelGGL(fused_softmax_kernel, dim3(N_ / 256, B_), dim3(256), 0, stream,
                     logits, target, keysA, acc);
  hipLaunchKernelGGL(seg_radix_sort_kernel, dim3(SEGS), dim3(256), 0, stream,
                     keysA, keysB);
  hipLaunchKernelGGL(lovasz_scan_kernel, dim3(SEGS), dim3(256), 0, stream,
                     keysA, acc, lossBC);
  hipLaunchKernelGGL(combine_kernel, dim3(1), dim3(32), 0, stream,
                     acc, lossBC, out);
}
